// TopKPerceptronRouter_77086073028657
// MI455X (gfx1250) — matrix operand order repaired and checked by reading, not compiled, not run on hardware
//
#include <hip/hip_runtime.h>
#include <hip/hip_bf16.h>

#define N_TOK 8192
#define D_DIM 4096
#define E_EXP 64
#define TOPK  2
#define WAVES 4
#define KCHUNK 32
#define NCHUNK (D_DIM / KCHUNK)
#define BSTRIDE 40  // ushorts per expert row in LDS (80 B: 16B-aligned, bank-spread)

typedef __attribute__((ext_vector_type(16))) __bf16         v16bf;
typedef __attribute__((ext_vector_type(16))) unsigned short v16u16;
typedef __attribute__((ext_vector_type(8)))  float          v8f;
#if __has_builtin(__builtin_amdgcn_cvt_pk_bf16_f32)
typedef __attribute__((ext_vector_type(2)))  __bf16         v2bf;
#endif

struct BPack { uint4 lo, hi; };  // 32 bytes -> one v16bf fragment

static __device__ __forceinline__ unsigned short f2bf(float f) {
  // round-to-nearest-even f32 -> bf16 (fallback path)
  unsigned int u = __float_as_uint(f);
  u += 0x7FFFu + ((u >> 16) & 1u);
  return (unsigned short)(u >> 16);
}

// Convert 8+8 fp32 (two K sub-ranges) into one packed bf16 A fragment.
static __device__ __forceinline__ v16bf pack_a(v8f a0, v8f a1) {
#if __has_builtin(__builtin_amdgcn_cvt_pk_bf16_f32)
  v16bf av;
#pragma unroll
  for (int i = 0; i < 4; ++i) {
    v2bf p0 = __builtin_amdgcn_cvt_pk_bf16_f32(a0[2 * i], a0[2 * i + 1]);
    v2bf p1 = __builtin_amdgcn_cvt_pk_bf16_f32(a1[2 * i], a1[2 * i + 1]);
    av[2 * i]     = p0[0];
    av[2 * i + 1] = p0[1];
    av[8 + 2 * i]     = p1[0];
    av[8 + 2 * i + 1] = p1[1];
  }
  return av;
#else
  v16u16 ap;
#pragma unroll
  for (int i = 0; i < 8; ++i) {
    ap[i]     = f2bf(a0[i]);
    ap[i + 8] = f2bf(a1[i]);
  }
  return __builtin_bit_cast(v16bf, ap);
#endif
}

// 32-bit LDS byte offset of a generic pointer into shared memory
typedef __attribute__((address_space(3))) const void lds_cvoid;
static __device__ __forceinline__ unsigned lds_off(const void* p) {
  return (unsigned)(unsigned long long)(lds_cvoid*)p;
}

// Async-stage one 4 KB B chunk (64 experts x 32 K bf16) into LDS.
// 128 threads x 2 x b128 = 4 KB; tracked on ASYNCcnt (2 per wave).
static __device__ __forceinline__ void stage_chunk(
    const unsigned short* __restrict__ Wbf, const unsigned short* ldsdst,
    int kc, int tid) {
  const int e    = tid >> 1;          // expert row this thread stages
  const int half = tid & 1;           // which 32 B half of the 64 B row
  const unsigned goff = (unsigned)(((e * D_DIM + kc) << 1) + (half << 5));
  const unsigned loff = lds_off(ldsdst + e * BSTRIDE + (half << 4));
  asm volatile(
      "global_load_async_to_lds_b128 %0, %1, %2\n\t"
      "global_load_async_to_lds_b128 %0, %1, %2 offset:16"
      :: "v"(loff), "v"(goff), "s"(Wbf) : "memory");
}

static __device__ __forceinline__ void wait_async(int keep2) {
#if __has_builtin(__builtin_amdgcn_s_wait_asynccnt)
  if (keep2) __builtin_amdgcn_s_wait_asynccnt(2);
  else       __builtin_amdgcn_s_wait_asynccnt(0);
#else
  if (keep2) asm volatile("s_wait_asynccnt 2" ::: "memory");
  else       asm volatile("s_wait_asynccnt 0" ::: "memory");
#endif
}

// One-time fp32 -> bf16 conversion of W into workspace (512 KB, L2-resident).
__global__ __launch_bounds__(256) void cvt_w_bf16(const float* __restrict__ W,
                                                  unsigned short* __restrict__ Wbf) {
  int i = blockIdx.x * blockDim.x + threadIdx.x;  // grid sized exactly to E*D
  Wbf[i] = f2bf(W[i]);
}

// Fused: logits = x @ W^T + b  ->  softmax  ->  top-2 (idx + prob).
// One wave owns a 16-row x 64-expert logits tile (4 WMMA column tiles).
// B chunks are double-buffered in LDS via async-to-LDS DMA shared by the block.
__global__ __launch_bounds__(32 * WAVES) void router_kernel(
    const float* __restrict__ x, const unsigned short* __restrict__ Wbf,
    const float* __restrict__ bias, int* __restrict__ idx_out,
    float* __restrict__ w_out) {
  __shared__ unsigned short Bs[2][E_EXP][BSTRIDE];  // 10 KB staging (2 buffers)
  __shared__ float slog[WAVES][16][68];             // logits, conflict-free scans

  const int tid  = threadIdx.x;
  const int lane = tid & 31;
  const int wv   = tid >> 5;
  const int row0 = (blockIdx.x * WAVES + wv) * 16;

  const int col = lane & 15;    // A row / B column within the 16x16 tile
  const int hi  = lane >> 4;    // lane half selects K sub-ranges
  const int kA  = hi ? 8  : 0;  // A: low half K 0..7 & 16..23; high 8..15 & 24..31
  const int kB  = hi ? 8  : 0;  // B: low half K 0..7 & 16..23; high 8..15 & 24..31

  const float* xrow = x + (size_t)(row0 + col) * D_DIM;

  v8f acc[4];
#pragma unroll
  for (int t = 0; t < 4; ++t) acc[t] = (v8f)(0.0f);

  stage_chunk(Wbf, &Bs[0][0][0], 0, tid);

  for (int c = 0; c < NCHUNK; ++c) {
    const int kc   = c * KCHUNK;
    const int cbuf = c & 1;

    __syncthreads();  // prior reads of the other buffer are done before overwrite
    const int more = (c + 1 < NCHUNK);
    if (more) stage_chunk(Wbf, &Bs[cbuf ^ 1][0][0], kc + KCHUNK, tid);

    // ---- A fragment: 16 fp32 per lane (streaming, read-once -> nontemporal)
    v8f a0 = __builtin_nontemporal_load((const v8f*)(xrow + kc + kA));
    v8f a1 = __builtin_nontemporal_load((const v8f*)(xrow + kc + kA + 16));
    __builtin_prefetch(xrow + kc + 64, 0, 1);  // gfx1250 global_prefetch_b8

    const v16bf av = pack_a(a0, a1);

    wait_async(more);   // current buffer's DMA complete (next chunk stays in flight)
    __syncthreads();    // ... for every wave in the block

    // ---- 4 expert tiles from LDS, then back-to-back WMMAs
    v16bf bv[4];
#pragma unroll
    for (int t = 0; t < 4; ++t) {
      const unsigned short* bp = &Bs[cbuf][t * 16 + col][kB];
      BPack pk;
      pk.lo = *(const uint4*)bp;
      pk.hi = *(const uint4*)(bp + 16);
      bv[t] = __builtin_bit_cast(v16bf, pk);
    }
#pragma unroll
    for (int t = 0; t < 4; ++t)
      acc[t] = __builtin_amdgcn_wmma_f32_16x16x32_bf16(
          false, av, false, bv[t], (short)0, acc[t], false, false);
  }

  // ---- Scatter C tiles to LDS: VGPR r holds M=r (lanes 0-15) / M=r+8 (16-31)
#pragma unroll
  for (int t = 0; t < 4; ++t)
#pragma unroll
    for (int r = 0; r < 8; ++r)
      slog[wv][r + 8 * hi][t * 16 + col] = acc[t][r];

  __syncthreads();

  // ---- Per-row softmax + top-2 (lanes 0-15 each own one row)
  if (lane < 16) {
    float m1 = -1e30f, m2 = -1e30f;
    int   i1 = 0,      i2 = 0;
#pragma unroll 4
    for (int e = 0; e < E_EXP; ++e) {
      float v = slog[wv][lane][e] + bias[e];
      slog[wv][lane][e] = v;
      if (v > m1)      { m2 = m1; i2 = i1; m1 = v; i1 = e; }
      else if (v > m2) { m2 = v;  i2 = e; }
    }
    float s = 0.0f;
#pragma unroll 4
    for (int e = 0; e < E_EXP; ++e) s += __expf(slog[wv][lane][e] - m1);

    const float p1 = 1.0f / s;              // exp(m1-m1)/s
    const float p2 = __expf(m2 - m1) / s;

    const int gr = row0 + lane;
    idx_out[gr * 2 + 0] = i1;
    idx_out[gr * 2 + 1] = i2;
    w_out[gr * 2 + 0]   = p1;
    w_out[gr * 2 + 1]   = p2;
  }
}

extern "C" void kernel_launch(void* const* d_in, const int* in_sizes, int n_in,
                              void* d_out, int out_size, void* d_ws, size_t ws_size,
                              hipStream_t stream) {
  const float* x = (const float*)d_in[0];   // [N, D] fp32
  const float* W = (const float*)d_in[1];   // [E, D] fp32
  const float* b = (const float*)d_in[2];   // [E]    fp32
  // d_in[3] = k (== 2), compile-time constant here

  unsigned short* Wbf = (unsigned short*)d_ws;           // 512 KB bf16 copy of W
  int*   idx_out = (int*)d_out;                          // [N, 2] int32 (first output)
  float* w_out   = (float*)d_out + (size_t)N_TOK * TOPK; // [N, 2] fp32  (second output)

  cvt_w_bf16<<<(E_EXP * D_DIM) / 256, 256, 0, stream>>>(W, Wbf);
  router_kernel<<<N_TOK / (16 * WAVES), 32 * WAVES, 0, stream>>>(
      x, Wbf, b, idx_out, w_out);
}
